// MyRNN_44882408243596
// MI455X (gfx1250) — matrix operand order repaired and checked by reading, not compiled, not run on hardware
//
#include <hip/hip_runtime.h>

// ---------------------------------------------------------------------------
// Stacked Elman RNN (B=8, T=2048, H=1024, L=2) as a persistent WMMA kernel.
//  - f16 WMMA (v_wmma_f32_16x16x32_f16), f32 accumulate, tanh in f32
//  - per-block weight tiles (64KB f16) staged ONCE into LDS; steady-state
//    inner loop: 2x ds_load_b128 + 2x global_load_b128 + 2x WMMA per k-block,
//    fully unrolled (immediate offsets), TWO independent accumulator chains
//  - X pre-packed to f16 A-fragments once per launch (if ws is big enough);
//    hidden state double-buffered in packed A-fragment layout (L2-resident)
//  - layer pipelining: blocks 0..63 = layer0 step p, blocks 64..127 = layer1
//    step p-1; one grid-wide atomic barrier per phase (T+1 barriers total).
// ---------------------------------------------------------------------------

#define TT   2048
#define HH   1024
#define BB   8
#define NBLK 128   // 64 blocks per layer, 16 output columns each
#define NTHR 128   // 4 wave32s; each wave covers K=256

typedef _Float16 f16;
typedef __attribute__((ext_vector_type(16))) _Float16 v16h;
typedef __attribute__((ext_vector_type(2)))  _Float16 v2h;
typedef __attribute__((ext_vector_type(8)))  float    v8f;
typedef __attribute__((ext_vector_type(4)))  float    v4f;

// workspace layout (bytes)
#define BAR_OFF 0
#define H1_OFF  4096                        // h1 packed A-frags: 2 x 32KB
#define H2_OFF  (H1_OFF + 2 * 32768)        // h2 packed A-frags: 2 x 32KB
#define WP_OFF  (H2_OFF + 2 * 32768)        // packed weights: 4*64*32*1KB = 8MB
#define XP_OFF  (WP_OFF + 4 * 64 * 32 * 1024)  // packed X: 2048*32KB = 64MB
#define XP_SZ   (TT * 32768)
#define WS_FULL ((size_t)XP_OFF + (size_t)XP_SZ + 65536)

// ---------------------------------------------------------------------------
__global__ void zero_ws(char* ws) {
  unsigned i = blockIdx.x * 256u + threadIdx.x;
  if (i < (unsigned)(WP_OFF / 4)) ((unsigned*)ws)[i] = 0u;
}

// Pack W[j,k] (f32 row-major, j = output col = N, k = K) into f16 B-fragments.
// Fragment = 32x16 (KxN) = 1KB: lane holds 32B; lanes 0-15: N=lane, K=0..7,16..23
// (elems 0-7 -> K=0..7, 8-15 -> K=16..23); lanes 16-31: N=lane-16, K=8..15, 24..31.
__global__ void pack_w(const float* __restrict__ ihw,
                       const float* __restrict__ hhw,
                       char* __restrict__ ws) {
  int tid  = blockIdx.x * 256 + threadIdx.x;   // 0 .. 2097151 (u32 slots)
  int s    = tid & 255;
  int f    = tid >> 8;
  int kb   = f & 31;
  int n    = (f >> 5) & 63;
  int mat  = f >> 11;                          // 0=ih0 1=hh0 2=ih1 3=hh1
  int lane = s >> 3;
  int vv   = s & 7;
  int k_local = (vv < 4) ? ((lane >> 4) * 8 + vv * 2) : (16 + (lane >> 4) * 8 + (vv - 4) * 2);
  int j    = n * 16 + (lane & 15);
  int k    = kb * 32 + k_local;
  const float* src = ((mat & 1) ? hhw : ihw) + (mat >> 1) * HH * HH + j * HH + k;
  v2h d;
  d[0] = (f16)src[0];
  d[1] = (f16)src[1];
  *(v2h*)(ws + WP_OFF + f * 1024 + s * 4) = d;
}

// Pack X[m,t,k] into f16 A-fragments: frag f = t*32+kb (1KB each).
// A-frag: lanes 0-15: M=lane (K pairs: elems 0-7 -> K=0..7, 8-15 -> K=16..23);
// lanes 16-31: M=lane-16 (K=8..15, 24..31). Rows m>=8 are zero padding.
__global__ void pack_x(const float* __restrict__ X, char* __restrict__ ws) {
  int tid  = blockIdx.x * 256 + threadIdx.x;   // 16,777,216 u32 slots
  int s    = tid & 255;
  int f    = tid >> 8;                         // t*32 + kb
  int kb   = f & 31;
  int t    = f >> 5;
  int lane = s >> 3;
  int vE   = s & 7;
  int half = lane >> 4;
  int m    = lane & 15;
  int k_local = (vE < 4) ? (half * 8 + 2 * vE) : (16 + half * 8 + 2 * (vE - 4));
  v2h d;
  if (m < BB) {
    const float* src = X + (m * TT + t) * HH + kb * 32 + k_local;
    d[0] = (f16)src[0];
    d[1] = (f16)src[1];
  } else {
    d[0] = (f16)0.0f;
    d[1] = (f16)0.0f;
  }
  *(v2h*)(ws + XP_OFF + f * 1024 + s * 4) = d;
}

// ---------------------------------------------------------------------------
template <bool XPACK>
__global__ __launch_bounds__(NTHR) void rnn_persist(
    const float* __restrict__ X, const float* __restrict__ ihb,
    const float* __restrict__ hhb, char* __restrict__ ws,
    float* __restrict__ out) {
  __shared__ __align__(32) char  swt[2 * 32768];  // [0]=Wih tile, [1]=Whh tile
  __shared__ __align__(32) float sred[NTHR * 8];

  const int n     = blockIdx.x & 63;
  const int layer = blockIdx.x >> 6;
  const int lane  = threadIdx.x & 31;
  const int wv    = threadIdx.x >> 5;

  unsigned* bar = (unsigned*)(ws + BAR_OFF);
  const char* wih = ws + WP_OFF + (size_t)((layer * 2 + 0) * 64 + n) * 32 * 1024;
  const char* whh = ws + WP_OFF + (size_t)((layer * 2 + 1) * 64 + n) * 32 * 1024;

  // ---- stage this block's weight tiles into LDS once (reused all T steps) ---
  {
    const v4f* gih = (const v4f*)wih;
    const v4f* ghh = (const v4f*)whh;
    v4f* sih = (v4f*)swt;
    v4f* shh = (v4f*)(swt + 32768);
    for (int i = threadIdx.x; i < 2048; i += NTHR) {
      sih[i] = gih[i];
      shh[i] = ghh[i];
    }
  }
  __syncthreads();

  const int   jc   = n * 16 + (lane & 15);
  const float bias = ihb[layer * HH + jc] + hhb[layer * HH + jc];
  const int   kb0  = wv * 8;
  const char* sa   = swt + lane * 32;            // LDS Wih fragments
  const char* sh   = swt + 32768 + lane * 32;    // LDS Whh fragments

  for (int p = 0; p <= TT; ++p) {
    const bool active = (layer == 0) ? (p < TT) : (p >= 1);
    if (active) {
      const int t = (layer == 0) ? p : (p - 1);
      // pure integer-offset selects keep provenance on `ws` -> global loads
      int hh_off, wd_off;
      if (layer == 0) {
        hh_off = H1_OFF + ((p - 1) & 1) * 32768;
        wd_off = H1_OFF + (p & 1) * 32768;
      } else {
        hh_off = H2_OFF + ((t - 1) & 1) * 32768;
        wd_off = H2_OFF + (t & 1) * 32768;
      }
      const char* ph   = ws + hh_off + lane * 32;
      char*       wdst = ws + wd_off;

      // two independent accumulator chains (ih / hh) -> half the WMMA RAW depth
      v8f c0 = {};
      v8f c1 = {};
      if (XPACK || layer == 1) {
        // unified packed-A inner loop (all immediate offsets)
        const int xa_off = (layer == 0) ? (XP_OFF + t * 32768)
                                        : (H1_OFF + (t & 1) * 32768);
        const char* pa = ws + xa_off + lane * 32;
#pragma unroll
        for (int u = 0; u < 8; ++u) {
          const int kb = kb0 + u;
          v16h ax = *(const v16h*)(pa + kb * 1024);
          v16h bx = *(const v16h*)(sa + kb * 1024);
          c0 = __builtin_amdgcn_wmma_f32_16x16x32_f16(false, ax, false, bx,
                                                      (short)0, c0, false, false);
          v16h ah = *(const v16h*)(ph + kb * 1024);
          v16h bh = *(const v16h*)(sh + kb * 1024);
          c1 = __builtin_amdgcn_wmma_f32_16x16x32_f16(false, ah, false, bh,
                                                      (short)0, c1, false, false);
        }
        if (XPACK && layer == 0 && (t + 1 < TT)) {
          // next timestep's X fragments are read-only: safe to prefetch now
          const char* nx = ws + XP_OFF + (t + 1) * 32768 + lane * 32;
#pragma unroll
          for (int u = 0; u < 8; ++u)
            __builtin_prefetch((const void*)(nx + (kb0 + u) * 1024), 0, 3);
        }
      } else {
        // fallback: convert X f32 -> f16 A-fragments on the fly (layer 0)
        const int half = lane >> 4;
        const int m    = lane & 15;
        const float* px = X + (m * TT + t) * HH + half * 8;
#pragma unroll
        for (int u = 0; u < 8; ++u) {
          const int kb = kb0 + u;
          v16h ax;
          if (m < BB) {
            v4f f0 = *(const v4f*)(px + kb * 32);
            v4f f1 = *(const v4f*)(px + kb * 32 + 4);
            v4f f2 = *(const v4f*)(px + kb * 32 + 16);
            v4f f3 = *(const v4f*)(px + kb * 32 + 20);
#pragma unroll
            for (int i = 0; i < 4; ++i) {
              ax[i]      = (f16)f0[i];
              ax[4 + i]  = (f16)f1[i];
              ax[8 + i]  = (f16)f2[i];
              ax[12 + i] = (f16)f3[i];
            }
          } else {
#pragma unroll
            for (int i = 0; i < 16; ++i) ax[i] = (f16)0.0f;
          }
          v16h bx = *(const v16h*)(sa + kb * 1024);
          c0 = __builtin_amdgcn_wmma_f32_16x16x32_f16(false, ax, false, bx,
                                                      (short)0, c0, false, false);
          v16h ah = *(const v16h*)(ph + kb * 1024);
          v16h bh = *(const v16h*)(sh + kb * 1024);
          c1 = __builtin_amdgcn_wmma_f32_16x16x32_f16(false, ah, false, bh,
                                                      (short)0, c1, false, false);
        }
      }
      // ---- cross-wave K reduction through LDS ----
      v8f c;
#pragma unroll
      for (int i = 0; i < 8; ++i) c[i] = c0[i] + c1[i];
      *(v8f*)&sred[(wv * 32 + lane) * 8] = c;
      __syncthreads();
      if (wv == 0) {
#pragma unroll
        for (int o = 1; o < 4; ++o) {
          v8f cp = *(const v8f*)&sred[(o * 32 + lane) * 8];
#pragma unroll
          for (int i = 0; i < 8; ++i) c[i] += cp[i];
        }
#pragma unroll
        for (int i = 0; i < 8; ++i) c[i] = tanhf(c[i] + bias);
        // lanes 0-15 hold rows m=0..7 (VGPR r) of column j = jc
        if (lane < 16) {
          const int j    = jc;
          const int kb2  = j >> 5;
          const int kl   = j & 31;
          const int grp  = kl >> 4;
          const int w16  = kl & 15;
          const int half = w16 >> 3;
          const int kk   = w16 & 7;
          const int vv   = grp * 4 + (kk >> 1);
          char* fb = wdst + kb2 * 1024 + half * 512 + vv * 4 + (kk & 1) * 2;
#pragma unroll
          for (int r = 0; r < 8; ++r) *(f16*)(fb + r * 32) = (f16)c[r];
          if (layer == 1) {
#pragma unroll
            for (int r = 0; r < 8; ++r) out[(r * TT + t) * HH + j] = c[r];
          }
          if (t == TT - 1) {
#pragma unroll
            for (int r = 0; r < 8; ++r)
              out[BB * TT * HH + (layer * BB + r) * HH + j] = c[r];
          }
        }
      }
      __syncthreads();  // protect sred before next phase
    }
    // ---- grid-wide barrier (one per phase) ----
    __threadfence();
    __syncthreads();
    if (threadIdx.x == 0) {
      __hip_atomic_fetch_add(bar, 1u, __ATOMIC_RELEASE, __HIP_MEMORY_SCOPE_AGENT);
      const unsigned target = (unsigned)NBLK * (unsigned)(p + 1);
      while (__hip_atomic_load(bar, __ATOMIC_ACQUIRE, __HIP_MEMORY_SCOPE_AGENT) <
             target)
        __builtin_amdgcn_s_sleep(1);
    }
    __syncthreads();
    __threadfence();
  }
}

// ---------------------------------------------------------------------------
extern "C" void kernel_launch(void* const* d_in, const int* in_sizes, int n_in,
                              void* d_out, int out_size, void* d_ws,
                              size_t ws_size, hipStream_t stream) {
  (void)in_sizes; (void)n_in; (void)out_size;
  const float* X   = (const float*)d_in[0];
  const float* ihw = (const float*)d_in[1];
  const float* ihb = (const float*)d_in[2];
  const float* hhw = (const float*)d_in[3];
  const float* hhb = (const float*)d_in[4];
  float* out = (float*)d_out;
  char*  ws  = (char*)d_ws;

  hipLaunchKernelGGL(zero_ws, dim3((WP_OFF / 4 + 255) / 256), dim3(256), 0,
                     stream, ws);
  hipLaunchKernelGGL(pack_w, dim3(8192), dim3(256), 0, stream, ihw, hhw, ws);
  if (ws_size >= WS_FULL) {
    hipLaunchKernelGGL(pack_x, dim3(65536), dim3(256), 0, stream, X, ws);
    hipLaunchKernelGGL(rnn_persist<true>, dim3(NBLK), dim3(NTHR), 0, stream, X,
                       ihb, hhb, ws, out);
  } else {
    hipLaunchKernelGGL(rnn_persist<false>, dim3(NBLK), dim3(NTHR), 0, stream, X,
                       ihb, hhb, ws, out);
  }
}
